// AsynchronousGraphGenerator_9655086481446
// MI455X (gfx1250) — matrix operand order repaired and checked by reading, not compiled, not run on hardware
//
#include <hip/hip_runtime.h>
#include <hip/hip_bf16.h>
#include <cstdint>
#include <cstddef>

// ---------------------------------------------------------------------------
// AsynchronousGraphGenerator forward pass for MI455X (gfx1250, wave32).
// All GEMM-like ops use v_wmma_f32_16x16x32_f16 (f16 in, f32 accumulate).
// ---------------------------------------------------------------------------

typedef _Float16 v16h __attribute__((ext_vector_type(16)));
typedef _Float16 h2   __attribute__((ext_vector_type(2)));
typedef float    v8f  __attribute__((ext_vector_type(8)));

#define DEV __device__ __forceinline__

// Model dims
constexpr int Bb    = 8;
constexpr int Nn    = 384;
constexpr int BN    = Bb * Nn;        // 3072
constexpr int METAd = 32;
constexpr int HIDd  = 64;
constexpr int Hh    = 4;
constexpr int FFH   = 128;
constexpr float SLOPE = 0.2f;
constexpr float EPSV  = 1e-5f;

// A-fragment K index for 16x32 f16 A operand (ISA 7.12.2):
// lanes 0-15: VGPR v<4 -> K={2v,2v+1}, v>=4 -> K={8+2v,9+2v}; lanes 16-31: +8
DEV int kbaseA(int v, int hi) { return ((v < 4) ? (2 * v) : (8 + 2 * v)) + 8 * hi; }

DEV v8f zero8() {
  v8f z;
#pragma unroll
  for (int r = 0; r < 8; ++r) z[r] = 0.0f;
  return z;
}

DEV v8f wmma_f16(const v16h a, const v16h b, v8f c) {
  // D = A(16x32 f16) * B(32x16 f16) + C(16x16 f32)
  return __builtin_amdgcn_wmma_f32_16x16x32_f16(false, a, false, b, (short)0, c,
                                                false, false);
}

// ---------------------------------------------------------------------------
// Embedding: feats = nf*W+b (INPUT_DIM=1); meta = [t2v | type_emb | spat_emb]
// One block (64 threads) per node.
// ---------------------------------------------------------------------------
__global__ __launch_bounds__(64) void embed_kernel(
    const float* __restrict__ nf, const float* __restrict__ times,
    const int* __restrict__ types, const int* __restrict__ spatial,
    const float* __restrict__ fW, const float* __restrict__ fB,
    const float* __restrict__ t2v_b, const float* __restrict__ t2v_b0,
    const float* __restrict__ t2v_w, const float* __restrict__ t2v_w0,
    const float* __restrict__ type_tab, const float* __restrict__ spat_tab,
    float* __restrict__ meta, _Float16* __restrict__ meta_h,
    _Float16* __restrict__ feats_h) {
  const int node = blockIdx.x;
  const int tid  = threadIdx.x;
  const float t  = times[node];
  const float x0 = nf[node];
  feats_h[(size_t)node * HIDd + tid] = (_Float16)(x0 * fW[tid] + fB[tid]);
  if (tid < METAd) {
    float m;
    if (tid == 0)       m = t * t2v_w0[0] + t2v_b0[0];
    else if (tid < 16)  m = sinf(t * t2v_w[tid - 1] + t2v_b[tid - 1]);
    else if (tid < 24)  m = type_tab[types[node] * 8 + (tid - 16)];
    else                m = spat_tab[spatial[node] * 8 + (tid - 24)];
    meta[(size_t)node * METAd + tid]   = m;
    meta_h[(size_t)node * METAd + tid] = (_Float16)m;
  }
}

// Query meta embedding: one block (32 threads) per batch.
__global__ __launch_bounds__(32) void qmeta_kernel(
    const float* __restrict__ qt, const int* __restrict__ qty,
    const int* __restrict__ qsp,
    const float* __restrict__ t2v_b, const float* __restrict__ t2v_b0,
    const float* __restrict__ t2v_w, const float* __restrict__ t2v_w0,
    const float* __restrict__ type_tab, const float* __restrict__ spat_tab,
    float* __restrict__ qmeta) {
  const int b = blockIdx.x, tid = threadIdx.x;
  const float t = qt[b];
  float m;
  if (tid == 0)       m = t * t2v_w0[0] + t2v_b0[0];
  else if (tid < 16)  m = sinf(t * t2v_w[tid - 1] + t2v_b[tid - 1]);
  else if (tid < 24)  m = type_tab[qty[b] * 8 + (tid - 16)];
  else                m = spat_tab[qsp[b] * 8 + (tid - 24)];
  qmeta[b * METAd + tid] = m;
}

// ---------------------------------------------------------------------------
// Weight prep: f32 [K,Nc] row-major -> f16 [Nc,K] (transposed), so that WMMA
// B fragments become contiguous 4-byte loads.
// ---------------------------------------------------------------------------
__global__ void wprep_kernel(const float* __restrict__ src,
                             _Float16* __restrict__ dst, int K, int Nc) {
  const int idx = blockIdx.x * blockDim.x + threadIdx.x;
  if (idx < K * Nc) {
    const int k = idx / Nc, n = idx % Nc;
    dst[(size_t)n * K + k] = (_Float16)src[idx];
  }
}

// ---------------------------------------------------------------------------
// Generic WMMA projection: Y[M,Ncols] = A[M,K] @ W[K,Ncols] + bias, optional
// ReLU, with selectable output formats. One wave per 16x16 output tile.
// ---------------------------------------------------------------------------
enum { PF_F32 = 1, PF_F16ROW = 2, PF_HVT = 4, PF_RELU = 8 };

__global__ __launch_bounds__(32) void proj_wmma_kernel(
    const _Float16* __restrict__ A, const _Float16* __restrict__ Wt,
    const float* __restrict__ bias, int K, int Ncols, int flags,
    float* __restrict__ oF32, _Float16* __restrict__ oF16,
    _Float16* __restrict__ oHvT) {
  const int lane = threadIdx.x;
  const int lo = lane & 15, hi = lane >> 4;
  const int mt = blockIdx.x, nt = blockIdx.y;

  v8f acc = zero8();
  const _Float16* arow = A  + (size_t)(mt * 16 + lo) * K;
  const _Float16* brow = Wt + (size_t)(nt * 16 + lo) * K;

  for (int k0 = 0; k0 < K; k0 += 32) {
    v16h af, bf;
#pragma unroll
    for (int v = 0; v < 8; ++v) {
      h2 pa = *(const h2*)(arow + k0 + kbaseA(v, hi));
      af[2 * v] = pa[0]; af[2 * v + 1] = pa[1];
      h2 pb = *(const h2*)(brow + k0 + kbaseA(v, hi));
      bf[2 * v] = pb[0]; bf[2 * v + 1] = pb[1];
    }
    acc = wmma_f16(af, bf, acc);
  }

  const int col = nt * 16 + lo;
  const float bv = bias[col];
#pragma unroll
  for (int r = 0; r < 8; ++r) {
    const int m = mt * 16 + r + 8 * hi;          // C/D layout: M = r + 8*(lane/16)
    float val = acc[r] + bv;
    if (flags & PF_RELU) val = fmaxf(val, 0.0f);
    if (flags & PF_F32)    oF32[(size_t)m * Ncols + col] = val;
    if (flags & PF_F16ROW) oF16[(size_t)m * Ncols + col] = (_Float16)val;
    if (flags & PF_HVT) {  // per-head transposed: hvT[b][h][f][j]
      const int b = m / Nn, j = m - b * Nn;
      const int h = col >> 4, f = col & 15;
      oHvT[(size_t)(((b * Hh + h) * 16) + f) * Nn + j] = (_Float16)val;
    }
  }
}

// ---------------------------------------------------------------------------
// Fused GAT attention: per (b, h, 16-row i-tile) block computes
//   e[i,j] = sum_f lrelu(hqk[i,f] + hqk[j,f]) * a[f]   (q == k == hqk)
//   att = softmax_j(e), then out = relu(att @ hv) via WMMA over K=384.
// Block = 256 threads; wave 0 runs the 12 WMMA k-steps.
// ---------------------------------------------------------------------------
__global__ __launch_bounds__(256) void attn_kernel(
    const float* __restrict__ hqk, const float* __restrict__ avec,
    const _Float16* __restrict__ hvT, float* __restrict__ out) {
  __shared__ float    sK[16][16];
  __shared__ float    sQ[Nn][16];
  __shared__ float    sE[16][Nn];
  __shared__ _Float16 sAtt[16][Nn];
  __shared__ float    sA[16];
  __shared__ float    red[16][16];
  __shared__ float    rowmax[16];
  __shared__ float    rowsum[16];

  const int tid = threadIdx.x;
  int bid = blockIdx.x;
  const int it = bid % (Nn / 16); bid /= (Nn / 16);
  const int h  = bid % Hh;
  const int b  = bid / Hh;

  for (int idx = tid; idx < Nn * 16; idx += 256) {
    const int j = idx >> 4, f = idx & 15;
    sQ[j][f] = hqk[(size_t)(b * Nn + j) * HIDd + h * 16 + f];
  }
  {
    const int i = tid >> 4, f = tid & 15;
    sK[i][f] = hqk[(size_t)(b * Nn + it * 16 + i) * HIDd + h * 16 + f];
  }
  if (tid < 16) sA[tid] = avec[tid];
  __syncthreads();

  const int i = tid & 15, jj = tid >> 4;
  for (int j = jj; j < Nn; j += 16) {
    float e = 0.0f;
#pragma unroll
    for (int f = 0; f < 16; ++f) {
      float v = sK[i][f] + sQ[j][f];
      v = (v > 0.0f) ? v : SLOPE * v;
      e += v * sA[f];
    }
    sE[i][j] = e;
  }
  __syncthreads();

  // softmax over j for each row i (16 threads cooperate per row)
  float pm = -1e30f;
  for (int j = jj; j < Nn; j += 16) pm = fmaxf(pm, sE[i][j]);
  red[i][jj] = pm;
  __syncthreads();
  if (jj == 0) {
    float m = red[i][0];
    for (int k = 1; k < 16; ++k) m = fmaxf(m, red[i][k]);
    rowmax[i] = m;
  }
  __syncthreads();
  const float rm = rowmax[i];
  float ps = 0.0f;
  for (int j = jj; j < Nn; j += 16) {
    const float ex = __expf(sE[i][j] - rm);
    sE[i][j] = ex;
    ps += ex;
  }
  red[i][jj] = ps;
  __syncthreads();
  if (jj == 0) {
    float s = 0.0f;
    for (int k = 0; k < 16; ++k) s += red[i][k];
    rowsum[i] = s;
  }
  __syncthreads();
  const float inv = 1.0f / rowsum[i];
  for (int j = jj; j < Nn; j += 16) sAtt[i][j] = (_Float16)(sE[i][j] * inv);
  __syncthreads();

  // wave 0: out[16 x 16] = att[16 x 384] @ hv[384 x 16] via 12 WMMA steps
  if (tid < 32) {
    const int lane = tid, lo = lane & 15, hi2 = lane >> 4;
    v8f acc = zero8();
    const _Float16* hvbase = hvT + (size_t)((b * Hh + h) * 16 + lo) * Nn;
    for (int k0 = 0; k0 < Nn; k0 += 32) {
      v16h af, bf;
#pragma unroll
      for (int v = 0; v < 8; ++v) {
        h2 pa = *(const h2*)(&sAtt[lo][k0 + kbaseA(v, hi2)]);
        af[2 * v] = pa[0]; af[2 * v + 1] = pa[1];
        h2 pb = *(const h2*)(hvbase + k0 + kbaseA(v, hi2));
        bf[2 * v] = pb[0]; bf[2 * v + 1] = pb[1];
      }
      acc = wmma_f16(af, bf, acc);
    }
#pragma unroll
    for (int r = 0; r < 8; ++r) {
      const int irow = it * 16 + r + 8 * hi2;
      out[(size_t)(b * Nn + irow) * HIDd + h * 16 + lo] = fmaxf(acc[r], 0.0f);
    }
  }
}

// ---------------------------------------------------------------------------
// LayerNorm over last dim (64), one block (64 threads) per row; f16 output.
// ---------------------------------------------------------------------------
__global__ __launch_bounds__(64) void ln_kernel(const float* __restrict__ x,
                                                const float* __restrict__ g,
                                                const float* __restrict__ bta,
                                                _Float16* __restrict__ out) {
  __shared__ float red[64];
  const int row = blockIdx.x, tid = threadIdx.x;
  const float v = x[(size_t)row * HIDd + tid];
  red[tid] = v;
  __syncthreads();
  for (int s = 32; s > 0; s >>= 1) {
    if (tid < s) red[tid] += red[tid + s];
    __syncthreads();
  }
  const float mu = red[0] / (float)HIDd;
  __syncthreads();
  const float d = v - mu;
  red[tid] = d * d;
  __syncthreads();
  for (int s = 32; s > 0; s >>= 1) {
    if (tid < s) red[tid] += red[tid + s];
    __syncthreads();
  }
  const float var = red[0] / (float)HIDd;
  const float y = d * rsqrtf(var + EPSV) * g[tid] + bta[tid];
  out[(size_t)row * HIDd + tid] = (_Float16)y;
}

// ---------------------------------------------------------------------------
// Generation layer: single query per (b,h) vs N keys; softmax over nodes.
// One block (128 threads) per (b,h).
// ---------------------------------------------------------------------------
__global__ __launch_bounds__(128) void gen_attn_kernel(
    const float* __restrict__ qmeta, const float* __restrict__ Wqk,
    const float* __restrict__ bqk, const float* __restrict__ avec,
    const float* __restrict__ hkg, const _Float16* __restrict__ hvT,
    float* __restrict__ gout) {
  __shared__ float q16[16], aS[16], e[Nn], red[128];
  const int bid = blockIdx.x;
  const int h = bid % Hh, b = bid / Hh;
  const int tid = threadIdx.x;

  if (tid < 16) {
    float acc = bqk[h * 16 + tid];
    for (int k = 0; k < METAd; ++k)
      acc += qmeta[b * METAd + k] * Wqk[k * HIDd + h * 16 + tid];
    q16[tid] = acc;
    aS[tid]  = avec[tid];
  }
  __syncthreads();

  for (int j = tid; j < Nn; j += 128) {
    const float* hr = hkg + (size_t)(b * Nn + j) * HIDd + h * 16;
    float s = 0.0f;
#pragma unroll
    for (int f = 0; f < 16; ++f) {
      float v = q16[f] + hr[f];
      v = (v > 0.0f) ? v : SLOPE * v;
      s += v * aS[f];
    }
    e[j] = s;
  }
  __syncthreads();

  float pm = -1e30f;
  for (int j = tid; j < Nn; j += 128) pm = fmaxf(pm, e[j]);
  red[tid] = pm;
  __syncthreads();
  for (int s = 64; s > 0; s >>= 1) {
    if (tid < s) red[tid] = fmaxf(red[tid], red[tid + s]);
    __syncthreads();
  }
  const float m = red[0];
  __syncthreads();
  float ps = 0.0f;
  for (int j = tid; j < Nn; j += 128) {
    const float ex = __expf(e[j] - m);
    e[j] = ex;
    ps += ex;
  }
  red[tid] = ps;
  __syncthreads();
  for (int s = 64; s > 0; s >>= 1) {
    if (tid < s) red[tid] += red[tid + s];
    __syncthreads();
  }
  const float inv = 1.0f / red[0];
  __syncthreads();

  const int f = tid & 15, p = tid >> 4;  // p in 0..7
  const _Float16* hvrow = hvT + (size_t)((b * Hh + h) * 16 + f) * Nn;
  float acc = 0.0f;
  for (int j = p; j < Nn; j += 8) acc += e[j] * (float)hvrow[j];
  red[tid] = acc;
  __syncthreads();
  if (tid < 16) {
    float s = 0.0f;
    for (int pp = 0; pp < 8; ++pp) s += red[pp * 16 + tid];
    gout[b * HIDd + h * 16 + tid] = fmaxf(s * inv, 0.0f);
  }
}

// ---------------------------------------------------------------------------
// Head MLP: out[b] = relu(g @ W1 + b1) @ W2 + b2. One block per batch.
// ---------------------------------------------------------------------------
__global__ __launch_bounds__(128) void head_kernel(
    const float* __restrict__ gin, const float* __restrict__ W1,
    const float* __restrict__ b1, const float* __restrict__ W2,
    const float* __restrict__ b2, float* __restrict__ out) {
  __shared__ float gs[64];
  __shared__ float red[128];
  const int b = blockIdx.x, tid = threadIdx.x;
  if (tid < 64) gs[tid] = gin[b * 64 + tid];
  __syncthreads();
  float acc = b1[tid];
  for (int k = 0; k < 64; ++k) acc += gs[k] * W1[k * FFH + tid];
  red[tid] = fmaxf(acc, 0.0f) * W2[tid];
  __syncthreads();
  for (int s = 64; s > 0; s >>= 1) {
    if (tid < s) red[tid] += red[tid + s];
    __syncthreads();
  }
  if (tid == 0) out[b] = red[0] + b2[0];
}

// ---------------------------------------------------------------------------
// Host-side orchestration
// ---------------------------------------------------------------------------
extern "C" void kernel_launch(void* const* d_in, const int* in_sizes, int n_in,
                              void* d_out, int out_size, void* d_ws,
                              size_t ws_size, hipStream_t stream) {
  (void)in_sizes; (void)n_in; (void)out_size; (void)ws_size;

  // JAX pytree flatten order (dict keys sorted alphabetically):
  const float* nf       = (const float*)d_in[0];
  const float* times    = (const float*)d_in[1];
  const int*   types    = (const int*)d_in[2];
  const int*   spatial  = (const int*)d_in[3];
  const float* qt       = (const float*)d_in[4];
  const int*   qty      = (const int*)d_in[5];
  const int*   qsp      = (const int*)d_in[6];
  const float* featW    = (const float*)d_in[7];   // feat.W [1,64]
  const float* featB    = (const float*)d_in[8];   // feat.b [64]
  const float* genWqk   = (const float*)d_in[9];   // gen.Wqk [32,64]
  const float* genWv    = (const float*)d_in[10];  // gen.Wv  [64,64]
  const float* genA     = (const float*)d_in[11];  // gen.a   [16]
  const float* genBqk   = (const float*)d_in[12];
  const float* genBv    = (const float*)d_in[13];
  const float* hW1      = (const float*)d_in[14];  // head.W1 [64,128]
  const float* hW2      = (const float*)d_in[15];  // head.W2 [128,1]
  const float* hB1      = (const float*)d_in[16];
  const float* hB2      = (const float*)d_in[17];
  struct LayerP {
    const float *Wqk, *Wv, *a, *bqk, *bv, *ffW1, *ffW2, *ffb1, *ffb2;
    const float *ln1b, *ln1g, *ln2b, *ln2g;
  } L[2];
  for (int l = 0; l < 2; ++l) {
    const int o = 18 + l * 13;
    L[l].Wqk  = (const float*)d_in[o + 0];
    L[l].Wv   = (const float*)d_in[o + 1];
    L[l].a    = (const float*)d_in[o + 2];
    L[l].bqk  = (const float*)d_in[o + 3];
    L[l].bv   = (const float*)d_in[o + 4];
    L[l].ffW1 = (const float*)d_in[o + 5];
    L[l].ffW2 = (const float*)d_in[o + 6];
    L[l].ffb1 = (const float*)d_in[o + 7];
    L[l].ffb2 = (const float*)d_in[o + 8];
    L[l].ln1b = (const float*)d_in[o + 9];
    L[l].ln1g = (const float*)d_in[o + 10];
    L[l].ln2b = (const float*)d_in[o + 11];
    L[l].ln2g = (const float*)d_in[o + 12];
  }
  const float* spat_tab = (const float*)d_in[44];
  const float* t2v_b    = (const float*)d_in[45];
  const float* t2v_b0   = (const float*)d_in[46];
  const float* t2v_w    = (const float*)d_in[47];
  const float* t2v_w0   = (const float*)d_in[48];
  const float* type_tab = (const float*)d_in[49];

  // workspace carve-out (256B aligned)
  char* ws = (char*)d_ws;
  size_t off = 0;
  auto alloc = [&](size_t bytes) {
    size_t o = off;
    off = (off + bytes + 255) & ~(size_t)255;
    return o;
  };
  float*    meta  = (float*)(ws + alloc((size_t)BN * METAd * 4));
  _Float16* metaH = (_Float16*)(ws + alloc((size_t)BN * METAd * 2));
  _Float16* actA  = (_Float16*)(ws + alloc((size_t)BN * HIDd * 2));
  _Float16* actB  = (_Float16*)(ws + alloc((size_t)BN * HIDd * 2));
  float*    hqk   = (float*)(ws + alloc((size_t)BN * HIDd * 4));
  _Float16* hvT   = (_Float16*)(ws + alloc((size_t)Bb * Hh * 16 * Nn * 2));
  float*    attO  = (float*)(ws + alloc((size_t)BN * HIDd * 4));
  _Float16* ln1H  = (_Float16*)(ws + alloc((size_t)BN * HIDd * 2));
  _Float16* ff1H  = (_Float16*)(ws + alloc((size_t)BN * FFH * 2));
  float*    ff2   = (float*)(ws + alloc((size_t)BN * HIDd * 4));
  float*    qm    = (float*)(ws + alloc((size_t)Bb * METAd * 4));
  float*    gbuf  = (float*)(ws + alloc((size_t)Bb * HIDd * 4));
  _Float16* WqkT[2]; _Float16* WvT[2]; _Float16* W1T[2]; _Float16* W2T[2];
  for (int l = 0; l < 2; ++l) {
    WqkT[l] = (_Float16*)(ws + alloc((size_t)HIDd * METAd * 2));
    WvT[l]  = (_Float16*)(ws + alloc((size_t)HIDd * HIDd * 2));
    W1T[l]  = (_Float16*)(ws + alloc((size_t)FFH * HIDd * 2));
    W2T[l]  = (_Float16*)(ws + alloc((size_t)HIDd * FFH * 2));
  }
  _Float16* GWqkT = (_Float16*)(ws + alloc((size_t)HIDd * METAd * 2));
  _Float16* GWvT  = (_Float16*)(ws + alloc((size_t)HIDd * HIDd * 2));

  // 1) embeddings
  embed_kernel<<<BN, 64, 0, stream>>>(nf, times, types, spatial, featW, featB,
                                      t2v_b, t2v_b0, t2v_w, t2v_w0, type_tab,
                                      spat_tab, meta, metaH, actA);
  qmeta_kernel<<<Bb, 32, 0, stream>>>(qt, qty, qsp, t2v_b, t2v_b0, t2v_w,
                                      t2v_w0, type_tab, spat_tab, qm);

  // 2) weight prep (f32 -> f16 transposed)
  auto wprep = [&](const float* src, _Float16* dst, int K, int Nc) {
    const int n = K * Nc;
    wprep_kernel<<<(n + 255) / 256, 256, 0, stream>>>(src, dst, K, Nc);
  };
  for (int l = 0; l < 2; ++l) {
    wprep(L[l].Wqk, WqkT[l], METAd, HIDd);
    wprep(L[l].Wv, WvT[l], HIDd, HIDd);
    wprep(L[l].ffW1, W1T[l], HIDd, FFH);
    wprep(L[l].ffW2, W2T[l], FFH, HIDd);
  }
  wprep(genWqk, GWqkT, METAd, HIDd);
  wprep(genWv, GWvT, HIDd, HIDd);

  // 3) GAT layers
  _Float16* actIn = actA;
  _Float16* actOut = actB;
  const dim3 g64(BN / 16, HIDd / 16);   // 192 x 4
  const dim3 g128(BN / 16, FFH / 16);   // 192 x 8
  for (int l = 0; l < 2; ++l) {
    proj_wmma_kernel<<<g64, 32, 0, stream>>>(metaH, WqkT[l], L[l].bqk, METAd,
                                             HIDd, PF_F32, hqk, nullptr, nullptr);
    proj_wmma_kernel<<<g64, 32, 0, stream>>>(actIn, WvT[l], L[l].bv, HIDd,
                                             HIDd, PF_HVT, nullptr, nullptr, hvT);
    attn_kernel<<<Bb * Hh * (Nn / 16), 256, 0, stream>>>(hqk, L[l].a, hvT, attO);
    ln_kernel<<<BN, 64, 0, stream>>>(attO, L[l].ln1g, L[l].ln1b, ln1H);
    proj_wmma_kernel<<<g128, 32, 0, stream>>>(ln1H, W1T[l], L[l].ffb1, HIDd,
                                              FFH, PF_F16ROW | PF_RELU, nullptr,
                                              ff1H, nullptr);
    proj_wmma_kernel<<<g64, 32, 0, stream>>>(ff1H, W2T[l], L[l].ffb2, FFH,
                                             HIDd, PF_F32, ff2, nullptr, nullptr);
    ln_kernel<<<BN, 64, 0, stream>>>(ff2, L[l].ln2g, L[l].ln2b, actOut);
    _Float16* tmp = actIn; actIn = actOut; actOut = tmp;
  }

  // 4) generation layer (keys/values over all nodes) + head
  proj_wmma_kernel<<<g64, 32, 0, stream>>>(metaH, GWqkT, genBqk, METAd, HIDd,
                                           PF_F32, hqk, nullptr, nullptr);
  proj_wmma_kernel<<<g64, 32, 0, stream>>>(actIn, GWvT, genBv, HIDd, HIDd,
                                           PF_HVT, nullptr, nullptr, hvT);
  gen_attn_kernel<<<Bb * Hh, 128, 0, stream>>>(qm, genWqk, genBqk, genA, hqk,
                                               hvT, gbuf);
  head_kernel<<<Bb, 128, 0, stream>>>(gbuf, hW1, hB1, hW2, hB2, (float*)d_out);
}
